// BertSelfAttention_7267084665023
// MI455X (gfx1250) — matrix operand order repaired and checked by reading, not compiled, not run on hardware
//
#include <hip/hip_runtime.h>
#include <cstdint>
#include <cstddef>

// Problem constants (B=2, S=2048, D=1024, H=16, HD=64)
#define S_LEN 2048
#define DIM   1024
#define NH    16
#define HD    64
#define BH    32      // B * NH

typedef __attribute__((ext_vector_type(16))) __bf16 v16bf;
typedef __attribute__((ext_vector_type(8)))  __bf16 v8bf;
typedef __attribute__((ext_vector_type(8)))  float  v8f;
typedef __attribute__((ext_vector_type(4)))  unsigned int u32x4;
typedef __attribute__((ext_vector_type(8)))  int i32x8;
typedef __attribute__((ext_vector_type(4)))  int i32x4;

// ---------------------------------------------------------------------------
// WMMA fragment helpers (wave32, 16x16x32 bf16).
// A-matrix 16x32 layout: lane<16 -> row=lane,    K = 0..7  and 16..23
//                        lane>=16 -> row=lane-16, K = 8..15 and 24..31
// B-matrix 32x16 layout: lane<16 -> col=lane,    K = 0..7  and 16..23
//                        lane>=16 -> col=lane-16, K = 8..15 and 24..31
// C/D 16x16 layout: VGPR r: lanes0-15 -> M=r, lanes16-31 -> M=r+8, N=lane%16
// ---------------------------------------------------------------------------
static __device__ inline v16bf comb(v8bf lo, v8bf hi) {
  return __builtin_shufflevector(lo, hi, 0,1,2,3,4,5,6,7,8,9,10,11,12,13,14,15);
}
// row = pointer to this lane's row (row length >= 32 bf16, 16B aligned)
static __device__ inline v16bf load_a_frag(const __bf16* row, int lane) {
  const v8bf* p = (const v8bf*)(row + ((lane & 16) ? 8 : 0));
  return comb(p[0], p[2]);   // {0..7,16..23} or {8..15,24..31}
}
static __device__ inline v16bf load_b_frag(const __bf16* row, int lane) {
  const v8bf* p = (const v8bf*)(row + ((lane & 16) ? 8 : 0));
  return comb(p[0], p[2]);   // {0..7,16..23} or {8..15,24..31}
}
static __device__ inline v8f wmma_bf16(v16bf a, v16bf b, v8f c) {
  return __builtin_amdgcn_wmma_f32_16x16x32_bf16(false, a, false, b, (short)0, c,
                                                 false, false);
}

// 16-lane butterfly reductions (rows of a C-tile live within a half-wave)
static __device__ inline float hmax16(float v) {
  for (int off = 8; off >= 1; off >>= 1) v = fmaxf(v, __shfl_xor(v, off, 32));
  return v;
}
static __device__ inline float hsum16(float v) {
  for (int off = 8; off >= 1; off >>= 1) v += __shfl_xor(v, off, 32);
  return v;
}

// ---------------------------------------------------------------------------
// Tensor Data Mover: build D# groups per CDNA5 ISA and
// issue TENSOR_LOAD_TO_LDS.  This toolchain exposes the 6-arg builtin:
//   (uint32x4 g0, int32x8 g1, int32x4 g2, int32x4 g3, int32x8 extra, i32 cpol)
// Groups 2/3 (and the extra group) are zero for <=2-D tensors.
// ---------------------------------------------------------------------------
#if defined(__HIP_DEVICE_COMPILE__) && __has_builtin(__builtin_amdgcn_tensor_load_to_lds)
#define USE_TDM 1
#else
#define USE_TDM 0
#endif

#if USE_TDM
static __device__ inline void tdm_load(unsigned lds_off, const void* g,
                                       unsigned tdim0, unsigned tdim1,
                                       unsigned tile0, unsigned tile1,
                                       unsigned stride0) {
  unsigned long long ga = (unsigned long long)(uintptr_t)g;
  u32x4 g0;
  g0[0] = 1u;                                      // count=1 (valid user D#)
  g0[1] = lds_off;                                 // LDS byte address
  g0[2] = (unsigned)ga;                            // global_addr[31:0]
  g0[3] = (unsigned)((ga >> 32) & 0x01FFFFFFull) | (2u << 30); // [56:32], type=2
  i32x8 g1;
  g1[0] = (int)(1u << 16);                         // data_size=1 -> 2 bytes/elem
  g1[1] = (int)((tdim0 & 0xFFFFu) << 16);          // tensor_dim0[15:0]
  g1[2] = (int)((tdim0 >> 16) | ((tdim1 & 0xFFFFu) << 16));
  g1[3] = (int)((tdim1 >> 16) | ((tile0 & 0xFFFFu) << 16));
  g1[4] = (int)(tile1 & 0xFFFFu);                  // tile_dim1 (tile_dim2 = 0)
  g1[5] = (int)stride0;                            // tensor_dim0_stride[31:0]
  g1[6] = 0;
  g1[7] = 0;
  i32x4 z4; z4[0] = z4[1] = z4[2] = z4[3] = 0;     // groups 2/3 unused (<=2D)
  i32x8 z8;
  for (int i = 0; i < 8; ++i) z8[i] = 0;
  __builtin_amdgcn_tensor_load_to_lds(g0, g1, z4, z4, z8, 0);
}
#endif

// ---------------------------------------------------------------------------
// Kernel 1: QKV projection.  out = X @ W^T + b, written as bf16 into
//   Q: [BH][S][HD]   (pre-scaled by 1/sqrt(HD))
//   K: [BH][S][HD]
//   V: [BH][HD][S]   (transposed, so flash B-fragments are contiguous)
// Block: 128 threads = 4 waves; block tile 128x64; wave tile 32x64 (2x4 WMMA).
// ---------------------------------------------------------------------------
__global__ __launch_bounds__(128) void qkv_proj_kernel(
    const float* __restrict__ X,
    const float* __restrict__ Wq, const float* __restrict__ Wk,
    const float* __restrict__ Wv,
    const float* __restrict__ bq, const float* __restrict__ bk,
    const float* __restrict__ bv,
    __bf16* __restrict__ qws, __bf16* __restrict__ kws,
    __bf16* __restrict__ vws) {
  constexpr int BM = 128, BN = 64, BK = 32;
  __shared__ __bf16 sA[BM * BK];
  __shared__ __bf16 sB[BN * BK];

  const int tid  = threadIdx.x;
  const int w    = tid >> 5;
  const int lane = tid & 31;
  const int z    = blockIdx.z;
  const float* W    = (z == 0) ? Wq : (z == 1) ? Wk : Wv;
  const float* bias = (z == 0) ? bq : (z == 1) ? bk : bv;
  const int m0 = blockIdx.x * BM;
  const int n0 = blockIdx.y * BN;

  v8f acc[2][4];
  for (int tm = 0; tm < 2; ++tm)
    for (int tn = 0; tn < 4; ++tn)
      for (int r = 0; r < 8; ++r) acc[tm][tn][r] = 0.f;

#pragma unroll 1
  for (int k0 = 0; k0 < DIM; k0 += BK) {
    // Stage A tile (128x32) and B tile (64x32) as bf16 into LDS.
    {
      const float4* xr = (const float4*)(X + (size_t)(m0 + tid) * DIM + k0);
      __bf16* da = sA + tid * BK;
#pragma unroll
      for (int i = 0; i < 8; ++i) {
        float4 f = xr[i];
        da[i * 4 + 0] = (__bf16)f.x; da[i * 4 + 1] = (__bf16)f.y;
        da[i * 4 + 2] = (__bf16)f.z; da[i * 4 + 3] = (__bf16)f.w;
      }
      if (tid < BN) {
        const float4* wr = (const float4*)(W + (size_t)(n0 + tid) * DIM + k0);
        __bf16* db = sB + tid * BK;
#pragma unroll
        for (int i = 0; i < 8; ++i) {
          float4 f = wr[i];
          db[i * 4 + 0] = (__bf16)f.x; db[i * 4 + 1] = (__bf16)f.y;
          db[i * 4 + 2] = (__bf16)f.z; db[i * 4 + 3] = (__bf16)f.w;
        }
      }
    }
    __syncthreads();

    v16bf af[2], bf[4];
#pragma unroll
    for (int tm = 0; tm < 2; ++tm)
      af[tm] = load_a_frag(sA + (w * 32 + tm * 16 + (lane & 15)) * BK, lane);
#pragma unroll
    for (int tn = 0; tn < 4; ++tn)
      bf[tn] = load_b_frag(sB + (tn * 16 + (lane & 15)) * BK, lane);
#pragma unroll
    for (int tm = 0; tm < 2; ++tm)
#pragma unroll
      for (int tn = 0; tn < 4; ++tn)
        acc[tm][tn] = wmma_bf16(af[tm], bf[tn], acc[tm][tn]);
    __syncthreads();
  }

  // Scatter-store with layout conversion into Q/K/V workspace.
#pragma unroll
  for (int tm = 0; tm < 2; ++tm) {
#pragma unroll
    for (int tn = 0; tn < 4; ++tn) {
      const int gcol = n0 + tn * 16 + (lane & 15);
      const float bv_ = bias[gcol];
      const int h = gcol >> 6, d = gcol & 63;
#pragma unroll
      for (int r = 0; r < 8; ++r) {
        const int grow = m0 + w * 32 + tm * 16 + r + ((lane & 16) ? 8 : 0);
        const int b = grow >> 11, s = grow & (S_LEN - 1);
        const float val = acc[tm][tn][r] + bv_;
        const size_t bh = (size_t)(b * NH + h);
        if (z == 0)
          qws[(bh * S_LEN + s) * HD + d] = (__bf16)(val * 0.125f); // 1/sqrt(64)
        else if (z == 1)
          kws[(bh * S_LEN + s) * HD + d] = (__bf16)val;
        else
          vws[(bh * HD + d) * S_LEN + s] = (__bf16)val;            // V^T
      }
    }
  }
}

// ---------------------------------------------------------------------------
// Kernel 2: flash attention.  grid = (S/64, BH); 128 threads = 4 waves.
// Each wave owns 16 query rows; K/V 64x64 tiles double-buffered in LDS via TDM.
// ---------------------------------------------------------------------------
__global__ __launch_bounds__(128) void attn_kernel(
    const __bf16* __restrict__ qws, const __bf16* __restrict__ kws,
    const __bf16* __restrict__ vws, const float* __restrict__ mask,
    float* __restrict__ out) {
  __shared__ __bf16 sK[2][64 * HD];   // [key][d]
  __shared__ __bf16 sV[2][HD * 64];   // [d][key]  (from V^T)
  __shared__ __bf16 sP[4][16 * 64];   // per-wave P staging

  const int tid  = threadIdx.x;
  const int w    = tid >> 5;
  const int lane = tid & 31;
  const int bh   = blockIdx.y;
  const int b    = bh >> 4;
  const int h    = bh & 15;
  const int q0   = blockIdx.x * 64;

  // Q fragments (A-matrix), loaded straight from global (already scaled).
  const __bf16* qrow =
      qws + ((size_t)bh * S_LEN + q0 + w * 16 + (lane & 15)) * HD;
  const v16bf qf0 = load_a_frag(qrow, lane);        // d = 0..31
  const v16bf qf1 = load_a_frag(qrow + 32, lane);   // d = 32..63

  v8f acc[4];
  float m_[8], l_[8];
  for (int n = 0; n < 4; ++n)
    for (int r = 0; r < 8; ++r) acc[n][r] = 0.f;
  for (int r = 0; r < 8; ++r) { m_[r] = -1e30f; l_[r] = 0.f; }

  const size_t kbase = (size_t)bh * S_LEN * HD;
  const size_t vbase = (size_t)bh * HD * S_LEN;

#if USE_TDM
  if (w == 0) {
    // K tile 0: contiguous 64*64 bf16 -> 1-D descriptor
    tdm_load((unsigned)(uintptr_t)&sK[0][0], kws + kbase,
             64 * HD, 1, 64 * HD, 0, 64 * HD);
    // V^T tile 0: 64 rows (d) x 64 keys, row stride S_LEN -> 2-D descriptor
    tdm_load((unsigned)(uintptr_t)&sV[0][0], vws + vbase,
             S_LEN, HD, 64, HD, S_LEN);
  }
#endif

#pragma unroll 1
  for (int j = 0; j < S_LEN / 64; ++j) {
#if USE_TDM
    if (w == 0) __builtin_amdgcn_s_wait_tensorcnt(0);
    __syncthreads();   // buffer j%2 ready for everyone; prior reads retired
    if (w == 0 && j + 1 < S_LEN / 64) {
      tdm_load((unsigned)(uintptr_t)&sK[(j + 1) & 1][0],
               kws + kbase + (size_t)(j + 1) * 64 * HD,
               64 * HD, 1, 64 * HD, 0, 64 * HD);
      tdm_load((unsigned)(uintptr_t)&sV[(j + 1) & 1][0],
               vws + vbase + (size_t)(j + 1) * 64,
               S_LEN, HD, 64, HD, S_LEN);
    }
    const __bf16* kb = &sK[j & 1][0];
    const __bf16* vb = &sV[j & 1][0];
#else
    __syncthreads();
    {
      const __bf16* gk = kws + kbase + (size_t)j * 64 * HD;
      const __bf16* gv = vws + vbase + (size_t)j * 64;
      for (int i = tid; i < 512; i += 128)
        ((v8bf*)&sK[0][0])[i] = ((const v8bf*)gk)[i];
      for (int i = tid; i < 512; i += 128) {
        const int row = i >> 3, c8 = i & 7;
        ((v8bf*)&sV[0][0])[i] = *(const v8bf*)(gv + (size_t)row * S_LEN + c8 * 8);
      }
    }
    __syncthreads();
    const __bf16* kb = &sK[0][0];
    const __bf16* vb = &sV[0][0];
#endif

    // ---- scores: S = Q @ K^T  (4 tiles of 16x16, K-dim = HD = 64) ----
    v8f st[4];
#pragma unroll
    for (int tn = 0; tn < 4; ++tn) {
      const __bf16* krow = kb + (tn * 16 + (lane & 15)) * HD;
      const v16bf kf0 = load_b_frag(krow, lane);
      const v16bf kf1 = load_b_frag(krow + 32, lane);
      v8f c;
      for (int r = 0; r < 8; ++r) c[r] = 0.f;
      c = wmma_bf16(qf0, kf0, c);
      c = wmma_bf16(qf1, kf1, c);
      st[tn] = c;
    }

    // ---- additive mask ----
    const int qr0 = q0 + w * 16 + ((lane & 16) ? 8 : 0);
#pragma unroll
    for (int tn = 0; tn < 4; ++tn) {
      const int kc = j * 64 + tn * 16 + (lane & 15);
      const float* mcol = mask + (size_t)b * S_LEN * S_LEN + kc;
#pragma unroll
      for (int r = 0; r < 8; ++r)
        st[tn][r] += mcol[(size_t)(qr0 + r) * S_LEN];
    }

    // ---- online softmax ----
#pragma unroll
    for (int r = 0; r < 8; ++r) {
      float v = fmaxf(fmaxf(st[0][r], st[1][r]), fmaxf(st[2][r], st[3][r]));
      v = hmax16(v);
      const float mn = fmaxf(m_[r], v);
      const float alpha = __expf(m_[r] - mn);
      m_[r] = mn;
      float rs = 0.f;
#pragma unroll
      for (int tn = 0; tn < 4; ++tn) {
        const float p = __expf(st[tn][r] - mn);
        st[tn][r] = p;
        rs += p;
      }
      rs = hsum16(rs);
      l_[r] = l_[r] * alpha + rs;
#pragma unroll
      for (int n = 0; n < 4; ++n) acc[n][r] *= alpha;
    }

    // ---- P: C-layout -> A-fragment layout via per-wave LDS staging ----
    __bf16* pw = &sP[w][0];
#pragma unroll
    for (int tn = 0; tn < 4; ++tn)
#pragma unroll
      for (int r = 0; r < 8; ++r)
        pw[(r + ((lane & 16) ? 8 : 0)) * 64 + tn * 16 + (lane & 15)] =
            (__bf16)st[tn][r];
    asm volatile("s_wait_dscnt 0" ::: "memory");
    const __bf16* prow = pw + (lane & 15) * 64;
    const v16bf pf0 = load_a_frag(prow, lane);        // keys 0..31
    const v16bf pf1 = load_a_frag(prow + 32, lane);   // keys 32..63

    // ---- O += P @ V  (4 output tiles along HD, K-dim = 64 keys) ----
#pragma unroll
    for (int n = 0; n < 4; ++n) {
      const __bf16* vrow = vb + (n * 16 + (lane & 15)) * 64;
      const v16bf vf0 = load_b_frag(vrow, lane);
      const v16bf vf1 = load_b_frag(vrow + 32, lane);
      acc[n] = wmma_bf16(pf0, vf0, acc[n]);
      acc[n] = wmma_bf16(pf1, vf1, acc[n]);
    }
  }

  // ---- finalize: O / l, write f32 ctx [B, S, H*HD] ----
  const int qrow0 = q0 + w * 16 + ((lane & 16) ? 8 : 0);
#pragma unroll
  for (int n = 0; n < 4; ++n) {
    const int col = h * HD + n * 16 + (lane & 15);
#pragma unroll
    for (int r = 0; r < 8; ++r)
      out[((size_t)b * S_LEN + qrow0 + r) * DIM + col] = acc[n][r] / l_[r];
  }
}

// ---------------------------------------------------------------------------
extern "C" void kernel_launch(void* const* d_in, const int* in_sizes, int n_in,
                              void* d_out, int out_size, void* d_ws,
                              size_t ws_size, hipStream_t stream) {
  const float* X    = (const float*)d_in[0];
  const float* mask = (const float*)d_in[1];
  const float* Wq   = (const float*)d_in[2];
  const float* bq   = (const float*)d_in[3];
  const float* Wk   = (const float*)d_in[4];
  const float* bk   = (const float*)d_in[5];
  const float* Wv   = (const float*)d_in[6];
  const float* bv   = (const float*)d_in[7];

  const size_t tsz = (size_t)BH * S_LEN * HD;   // elements per tensor
  __bf16* qws = (__bf16*)d_ws;
  __bf16* kws = qws + tsz;
  __bf16* vws = kws + tsz;

  qkv_proj_kernel<<<dim3((2 * S_LEN) / 128, DIM / 64, 3), 128, 0, stream>>>(
      X, Wq, Wk, Wv, bq, bk, bv, qws, kws, vws);
  attn_kernel<<<dim3(S_LEN / 64, BH), 128, 0, stream>>>(qws, kws, vws, mask,
                                                        (float*)d_out);
}
